// CMX_Block_4784593568307
// MI455X (gfx1250) — matrix operand order repaired and checked by reading, not compiled, not run on hardware
//
#include <hip/hip_runtime.h>
#include <cstddef>

// ---------------------------------------------------------------------------
// Types for WMMA / TDM (gfx1250, wave32). Trivially-constructible ext-vectors
// so they can live in unions and match the probed builtin signatures.
// ---------------------------------------------------------------------------
typedef float v8f  __attribute__((ext_vector_type(8)));
typedef __bf16 v16bf __attribute__((ext_vector_type(16)));
typedef unsigned int u32x4 __attribute__((ext_vector_type(4)));
typedef int i32x4 __attribute__((ext_vector_type(4)));
typedef int i32x8 __attribute__((ext_vector_type(8)));

union U16 { u32x4 q[2]; v16bf v; unsigned short s[16]; };
union U8  { u32x4 q; unsigned short s[8]; };

__device__ __forceinline__ float bf2f(unsigned short h) {
  return __uint_as_float(((unsigned int)h) << 16);
}
__device__ __forceinline__ unsigned short f2bf(float f) {
  unsigned int u = __float_as_uint(f);
  u += 0x7FFFu + ((u >> 16) & 1u);            // round-to-nearest-even
  return (unsigned short)(u >> 16);
}

// ---------------------------------------------------------------------------
// Tensor Data Mover: 2D tile load Global->LDS (ISA 08_async_tensor §8).
// D# group0: count=1, lds_addr, global_addr (tile start), type=2.
// D# group1: data_size, tensor_dim0/1 (OOB bounds; OOB rows read as zero),
//            tile_dim0/1, tensor_dim0_stride.
// data_size_code: 0=1B 1=2B 2=4B.  Issued by one wave; tracked on TENSORcnt.
// ---------------------------------------------------------------------------
__device__ __forceinline__ void tdm_load_2d(unsigned lds_off, const void* gptr,
                                            unsigned tile_w, unsigned tile_h,
                                            unsigned stride_elems,
                                            unsigned rows_valid,
                                            unsigned data_size_code) {
  unsigned long long ga = (unsigned long long)(size_t)gptr;
  u32x4 g0 = {1u, lds_off, (unsigned)ga,
              (unsigned)((ga >> 32) & 0x01FFFFFFu) | (2u << 30)};
  unsigned td0 = tile_w;                 // tensor_dim0 (x OOB bound)
  unsigned td1 = rows_valid;             // tensor_dim1 (y OOB bound -> zero fill)
  i32x8 g1;
  g1[0] = (int)(data_size_code << 16);                       // mask=0, flags=0
  g1[1] = (int)((td0 & 0xFFFFu) << 16);                      // dim0[15:0] @48
  g1[2] = (int)((td0 >> 16) | ((td1 & 0xFFFFu) << 16));      // dim0 hi | dim1 lo
  g1[3] = (int)((td1 >> 16) | ((tile_w & 0xFFFFu) << 16));   // dim1 hi | tile0
  g1[4] = (int)(tile_h & 0xFFFFu);                           // tile1 | tile2=0
  g1[5] = (int)stride_elems;                                 // dim0_stride lo
  g1[6] = 0;
  g1[7] = 0;
  i32x4 gz = {0, 0, 0, 0};
#if defined(__clang_major__) && (__clang_major__ >= 23)
  i32x8 gz8 = {0, 0, 0, 0, 0, 0, 0, 0};
  __builtin_amdgcn_tensor_load_to_lds(g0, g1, gz, gz, gz8, 0);
#else
  __builtin_amdgcn_tensor_load_to_lds(g0, g1, gz, gz, 0);
#endif
}

__device__ __forceinline__ unsigned lds_addr_of(const void* p) {
  // Generic (flat) LDS pointers carry the workgroup-relative byte offset in
  // their low 32 bits (ISA §10.2 aperture mapping).
  return (unsigned)(size_t)p;
}

// ---------------------------------------------------------------------------
// Problem constants
// ---------------------------------------------------------------------------
constexpr int B_   = 4;
constexpr int HW   = 64;
constexpr int C_   = 768;
constexpr int NHED = 12;
constexpr int HD   = 64;
constexpr int WIN_ = 14;
constexpr int NWIN = B_ * 5 * 5;          // 100 windows
constexpr int NTOK = WIN_ * WIN_;         // 196 tokens / window
constexpr int MW   = NWIN * NTOK;         // 19600 window-token rows
constexpr int NP   = 224;                 // 196 padded to 7*32 for WMMA K-loop
constexpr int MT   = B_ * HW * HW;        // 16384 tokens
constexpr int DFF  = 3072;

// ---------------------------------------------------------------------------
// Workspace layout (bytes).  Overlays exploit lifetimes:
//   Q region  -> later h_ln2 (bf16 16384x768)
//   K region  -> h_mid start (bf16 16384x3072 spans K+V+AFEAT)
//   AWIN      -> later attn_out, later s_relu
// Total ~301 MB.
// ---------------------------------------------------------------------------
constexpr size_t AL(size_t x) { return (x + 255) & ~(size_t)255; }
constexpr size_t SZ_QKV1   = (size_t)NWIN * NHED * NP * HD * 2;   // 34,406,400
constexpr size_t OFF_WQKV  = 0;
constexpr size_t OFF_WPROJ = OFF_WQKV  + AL((size_t)3 * C_ * C_ * 2);
constexpr size_t OFF_WFC1  = OFF_WPROJ + AL((size_t)C_ * C_ * 2);
constexpr size_t OFF_WFC2  = OFF_WFC1  + AL((size_t)DFF * C_ * 2);
constexpr size_t OFF_WSW1  = OFF_WFC2  + AL((size_t)C_ * DFF * 2);
constexpr size_t OFF_Q     = OFF_WSW1  + AL((size_t)C_ * 2 * C_ * 2);
constexpr size_t OFF_K     = OFF_Q + SZ_QKV1;
constexpr size_t OFF_V     = OFF_K + SZ_QKV1;
constexpr size_t OFF_AFEAT = OFF_V + SZ_QKV1;                      // bf16 16384x1536
constexpr size_t OFF_AWIN  = OFF_AFEAT + AL((size_t)MT * 2 * C_ * 2);
constexpr size_t SZ_AWIN   = (size_t)MW * C_ * 2;                  // 30,105,600
constexpr size_t OFF_XATTN = OFF_AWIN + AL(SZ_AWIN);               // f32 16384x768
constexpr size_t OFF_XFIN  = OFF_XATTN + AL((size_t)MT * C_ * 4);
constexpr size_t OFF_YVEC  = OFF_XFIN + AL((size_t)MT * C_ * 4);
constexpr size_t OFF_T1    = OFF_YVEC + AL((size_t)B_ * 4 * C_ * 4);
constexpr size_t OFF_CWV   = OFF_T1   + AL((size_t)B_ * 4 * C_ * 4);
constexpr size_t OFF_S2    = OFF_CWV  + AL((size_t)B_ * 2 * C_ * 4);
// aliases
constexpr size_t OFF_HLN2  = OFF_Q;      // bf16 16384x768  (q dead by then)
constexpr size_t OFF_HMID  = OFF_K;      // bf16 16384x3072 (k,v,afeat dead)
constexpr size_t OFF_AOUT  = OFF_AWIN;   // bf16 19600x768  (A_win dead)
constexpr size_t OFF_SRELU = OFF_AWIN;   // bf16 16384x768  (attn_out dead)

// ---------------------------------------------------------------------------
// f32 -> bf16 conversion kernel (weights)
// ---------------------------------------------------------------------------
__global__ __launch_bounds__(256) void cvt_bf16(const float* __restrict__ in,
                                                unsigned short* __restrict__ out,
                                                int n) {
  int i = blockIdx.x * 256 + threadIdx.x;
  if (i < n) out[i] = f2bf(in[i]);
}

// ---------------------------------------------------------------------------
// LayerNorm (block per token, 256 threads, 768 channels).
// mode 1: scatter into window-gathered A_win layout; mode 0: plain rows.
// ---------------------------------------------------------------------------
__global__ __launch_bounds__(256) void ln_kernel(const float* __restrict__ x,
                                                 const float* __restrict__ w,
                                                 const float* __restrict__ b,
                                                 unsigned short* __restrict__ out,
                                                 int window_gather) {
  __shared__ float red[256];
  const int t = blockIdx.x;
  const float* xr = x + (size_t)t * C_;
  float v0[3], s = 0.f;
#pragma unroll
  for (int i = 0; i < 3; ++i) { v0[i] = xr[threadIdx.x + i * 256]; s += v0[i]; }
  red[threadIdx.x] = s; __syncthreads();
  for (int off = 128; off > 0; off >>= 1) {
    if (threadIdx.x < off) red[threadIdx.x] += red[threadIdx.x + off];
    __syncthreads();
  }
  const float mean = red[0] * (1.0f / 768.0f);
  __syncthreads();
  float s2 = 0.f;
#pragma unroll
  for (int i = 0; i < 3; ++i) { float d = v0[i] - mean; s2 += d * d; }
  red[threadIdx.x] = s2; __syncthreads();
  for (int off = 128; off > 0; off >>= 1) {
    if (threadIdx.x < off) red[threadIdx.x] += red[threadIdx.x + off];
    __syncthreads();
  }
  const float rs = rsqrtf(red[0] * (1.0f / 768.0f) + 1e-6f);
  size_t orow;
  if (window_gather) {
    int bb = t >> 12, hh = (t >> 6) & 63, ww = t & 63;
    int win = bb * 25 + (hh / WIN_) * 5 + (ww / WIN_);
    orow = (size_t)win * NTOK + (hh % WIN_) * WIN_ + (ww % WIN_);
  } else {
    orow = (size_t)t;
  }
#pragma unroll
  for (int i = 0; i < 3; ++i) {
    int c = threadIdx.x + i * 256;
    out[orow * C_ + c] = f2bf((v0[i] - mean) * rs * w[c] + b[c]);
  }
}

// ---------------------------------------------------------------------------
// Generic bf16 WMMA GEMM:  Out = A(MxK) @ W(NxK)^T  with fused epilogue.
// 256 threads = 8 waves in 4(M) x 2(N); 128x128 tile, K-step 32.
// Tiles staged by the Tensor Data Mover, double-buffered: wave 0 issues the
// DMA for step s+1 right after the barrier, so it overlaps the WMMAs of step
// s; one s_barrier per K-step.  TDM OOB (row >= tensor_dim1) zero-fills, which
// implements the ragged-M guard for free.
// ---------------------------------------------------------------------------
struct EpiParams {
  const float* bias;
  const float* extra;        // shortcut input (proj / fc2)
  float* outf;
  unsigned short* outb;
  unsigned short* oq;
  unsigned short* ok;
  unsigned short* ov;
};

template <int MODE>
__device__ __forceinline__ void epi(int gm, int gn, float v, const EpiParams& p) {
  v += p.bias[gn];
  if constexpr (MODE == 0) {            // QKV scatter, q pre-scaled by hd^-1/2
    int win = gm / NTOK, n = gm - win * NTOK;
    int t = gn / C_;  int rem = gn - t * C_;
    int head = rem >> 6, d = rem & 63;
    size_t dst = ((size_t)(win * NHED + head) * NP + n) * HD + d;
    if (t == 0)      p.oq[dst] = f2bf(v * 0.125f);
    else if (t == 1) p.ok[dst] = f2bf(v);
    else             p.ov[dst] = f2bf(v);
  } else if constexpr (MODE == 1) {     // proj + shortcut, window -> (b,h,w)
    int win = gm / NTOK, n = gm - win * NTOK;
    int bb = win / 25, wy = (win - bb * 25) / 5, wx = win % 5;
    int hh = wy * WIN_ + n / WIN_, ww = wx * WIN_ + n % WIN_;
    if (hh < HW && ww < HW) {
      size_t idx = ((size_t)((bb * HW + hh) * HW + ww)) * C_ + gn;
      p.outf[idx] = p.extra[idx] + v;
    }
  } else if constexpr (MODE == 2) {     // fc1 + exact GELU -> bf16
    float g = 0.5f * v * (1.0f + erff(v * 0.70710678118654752f));
    p.outb[(size_t)gm * DFF + gn] = f2bf(g);
  } else if constexpr (MODE == 3) {     // fc2 + residual -> f32
    size_t idx = (size_t)gm * C_ + gn;
    p.outf[idx] = p.extra[idx] + v;
  } else {                              // sw1 + ReLU -> bf16
    p.outb[(size_t)gm * C_ + gn] = f2bf(v > 0.f ? v : 0.f);
  }
}

template <int MODE>
__global__ __launch_bounds__(256) void gemm_bf16(
    const unsigned short* __restrict__ A, const unsigned short* __restrict__ W,
    int M, int N, int K, EpiParams p) {
  __shared__ __align__(16) unsigned short AsB[2][128 * 32];
  __shared__ __align__(16) unsigned short BsB[2][128 * 32];
  const int tid = threadIdx.x, lane = tid & 31, wave = tid >> 5;
  const int wm = wave >> 1, wn = wave & 1;
  const int halfsel = lane >> 4, lrow = lane & 15;
  const int bm = blockIdx.y * 128, bn = blockIdx.x * 128;
  const int steps = K >> 5;
  const unsigned rows_valid = (unsigned)((M - bm) < 128 ? (M - bm) : 128);

  v8f acc[2][4];
  const v8f vz = {0.f, 0.f, 0.f, 0.f, 0.f, 0.f, 0.f, 0.f};
#pragma unroll
  for (int i = 0; i < 2; ++i)
#pragma unroll
    for (int j = 0; j < 4; ++j) acc[i][j] = vz;

  const unsigned as_off[2] = {lds_addr_of(&AsB[0][0]), lds_addr_of(&AsB[1][0])};
  const unsigned bs_off[2] = {lds_addr_of(&BsB[0][0]), lds_addr_of(&BsB[1][0])};

  if (wave == 0) {                      // prologue: DMA tiles for step 0
    tdm_load_2d(as_off[0], A + (size_t)bm * K, 32, 128, (unsigned)K, rows_valid, 1);
    tdm_load_2d(bs_off[0], W + (size_t)bn * K, 32, 128, (unsigned)K, 128, 1);
  }

  for (int s = 0; s < steps; ++s) {
    if (wave == 0) __builtin_amdgcn_s_wait_tensorcnt(0);
    __syncthreads();                    // tiles for step s visible; prior reads done
    if (wave == 0 && s + 1 < steps) {   // overlap next DMA with this step's WMMAs
      tdm_load_2d(as_off[(s + 1) & 1], A + (size_t)bm * K + (size_t)(s + 1) * 32,
                  32, 128, (unsigned)K, rows_valid, 1);
      tdm_load_2d(bs_off[(s + 1) & 1], W + (size_t)bn * K + (size_t)(s + 1) * 32,
                  32, 128, (unsigned)K, 128, 1);
    }
    const unsigned short* As = AsB[s & 1];
    const unsigned short* Bs = BsB[s & 1];
    U16 a[2], bop[4];
#pragma unroll
    for (int sm = 0; sm < 2; ++sm) {                 // A operand: ISA 16x32 layout
      int r = wm * 32 + sm * 16 + lrow;
      a[sm].q[0] = *(const u32x4*)(As + r * 32 + halfsel * 8);
      a[sm].q[1] = *(const u32x4*)(As + r * 32 + 16 + halfsel * 8);
    }
#pragma unroll
    for (int sn = 0; sn < 4; ++sn) {                 // B operand: 32x16 layout
      int n = wn * 64 + sn * 16 + lrow;
      bop[sn].q[0] = *(const u32x4*)(Bs + n * 32 + halfsel * 8);
      bop[sn].q[1] = *(const u32x4*)(Bs + n * 32 + 16 + halfsel * 8);
    }
#pragma unroll
    for (int sm = 0; sm < 2; ++sm)
#pragma unroll
      for (int sn = 0; sn < 4; ++sn)
        acc[sm][sn] = __builtin_amdgcn_wmma_f32_16x16x32_bf16(
            false, a[sm].v, false, bop[sn].v, (short)0, acc[sm][sn], false, false);
  }
#pragma unroll
  for (int sm = 0; sm < 2; ++sm)
#pragma unroll
    for (int sn = 0; sn < 4; ++sn) {
      int gmb = bm + wm * 32 + sm * 16 + halfsel * 8;
      int gnn = bn + wn * 64 + sn * 16 + lrow;
#pragma unroll
      for (int r = 0; r < 8; ++r) {
        int gm = gmb + r;
        if (gm < M) epi<MODE>(gm, gnn, acc[sm][sn][r], p);
      }
    }
}

// ---------------------------------------------------------------------------
// Fused windowed attention: one block per (window, head).
// LDS: q(224x64 bf16) k(224x64) vT(64x224) relH/relW(27x64 f32) S(224x224 f32)
// q, k and the rel-pos tables are fetched by the TDM; v is transposed by the
// waves (TDM cannot transpose).
// ---------------------------------------------------------------------------
constexpr int ATTN_SMEM = 3 * NP * HD * 2 + 2 * 27 * HD * 4 + NP * NP * 4; // 300544

__global__ __launch_bounds__(256) void attn_kernel(
    const unsigned short* __restrict__ qbuf, const unsigned short* __restrict__ kbuf,
    const unsigned short* __restrict__ vbuf, const float* __restrict__ rel_h,
    const float* __restrict__ rel_w, unsigned short* __restrict__ attn_out) {
  extern __shared__ __align__(16) char smem[];
  unsigned short* qs  = (unsigned short*)smem;       // 224*64
  unsigned short* ksm = qs + NP * HD;                // 224*64
  unsigned short* vts = ksm + NP * HD;               // 64*224 (transposed)
  float* rhs = (float*)(smem + 3 * NP * HD * 2);     // 27*64
  float* rws = rhs + 27 * HD;
  float* S   = rws + 27 * HD;                        // 224*224

  const int blk = blockIdx.x;                        // win*12 + head
  const int tid = threadIdx.x, lane = tid & 31, wave = tid >> 5;
  const int halfsel = lane >> 4, lrow = lane & 15;
  const size_t base = (size_t)blk * NP * HD;

  if (wave == 0) {                                   // TDM: q, k, rel tables
    tdm_load_2d(lds_addr_of(qs),  qbuf + base, NP * HD, 1, NP * HD, 1, 1);
    tdm_load_2d(lds_addr_of(ksm), kbuf + base, NP * HD, 1, NP * HD, 1, 1);
    tdm_load_2d(lds_addr_of(rhs), rel_h, 27 * HD, 1, 27 * HD, 1, 2);
    tdm_load_2d(lds_addr_of(rws), rel_w, 27 * HD, 1, 27 * HD, 1, 2);
  }
  for (int i = tid; i < NP * HD / 8; i += 256) {     // v transposed -> [d][tok]
    U8 u; u.q = ((const u32x4*)(vbuf + base))[i];
    int tok = i >> 3, d0 = (i & 7) * 8;
#pragma unroll
    for (int j = 0; j < 8; ++j) vts[(d0 + j) * NP + tok] = u.s[j];
  }
  if (wave == 0) __builtin_amdgcn_s_wait_tensorcnt(0);
  __syncthreads();

  // ---- S = (q*scale) @ k^T via WMMA: 14x14 tiles, K = 64 in two steps ----
  for (int t = wave; t < 196; t += 8) {
    int tr = t / 14, tc = t % 14;
    v8f c = {0.f, 0.f, 0.f, 0.f, 0.f, 0.f, 0.f, 0.f};
#pragma unroll
    for (int ks0 = 0; ks0 < HD; ks0 += 32) {
      U16 a, b;
      int qr = tr * 16 + lrow;
      a.q[0] = *(const u32x4*)(qs + qr * HD + ks0 + halfsel * 8);
      a.q[1] = *(const u32x4*)(qs + qr * HD + ks0 + 16 + halfsel * 8);
      int kr = tc * 16 + lrow;
      b.q[0] = *(const u32x4*)(ksm + kr * HD + ks0 + halfsel * 8);
      b.q[1] = *(const u32x4*)(ksm + kr * HD + ks0 + 16 + halfsel * 8);
      c = __builtin_amdgcn_wmma_f32_16x16x32_bf16(false, a.v, false, b.v,
                                                  (short)0, c, false, false);
    }
#pragma unroll
    for (int r = 0; r < 8; ++r)
      S[(size_t)(tr * 16 + r + halfsel * 8) * NP + tc * 16 + lrow] = c[r];
  }
  __syncthreads();

  // ---- rel-pos bias + softmax (f32, per query row) ----
  if (tid < 196) {
    const int row = tid, qh = row / 14, qw = row % 14;
    float qv[HD];
#pragma unroll
    for (int d = 0; d < HD; ++d) qv[d] = bf2f(qs[row * HD + d]) * 8.0f; // unscale
    float rh[14], rw[14];
    for (int kk = 0; kk < 14; ++kk) {
      const float* ph = rhs + (qh - kk + 13) * HD;
      const float* pw = rws + (qw - kk + 13) * HD;
      float sh = 0.f, sw = 0.f;
#pragma unroll
      for (int d = 0; d < HD; ++d) { sh += qv[d] * ph[d]; sw += qv[d] * pw[d]; }
      rh[kk] = sh; rw[kk] = sw;
    }
    float* Srow = S + (size_t)row * NP;
    float mx = -1e30f;
    for (int kh = 0; kh < 14; ++kh)
      for (int kw = 0; kw < 14; ++kw) {
        int nk = kh * 14 + kw;
        float v = Srow[nk] + rh[kh] + rw[kw];
        Srow[nk] = v;
        mx = fmaxf(mx, v);
      }
    float sum = 0.f;
    for (int nk = 0; nk < 196; ++nk) {
      float e = __expf(Srow[nk] - mx);
      Srow[nk] = e; sum += e;
    }
    float inv = 1.0f / sum;
    for (int nk = 0; nk < 196; ++nk) Srow[nk] *= inv;
    for (int nk = 196; nk < NP; ++nk) Srow[nk] = 0.f;   // kill padded keys
  }
  __syncthreads();

  // ---- O = P @ V via WMMA: 14x4 tiles, K = 224 in seven steps ----
  const int win = blk / NHED, head = blk % NHED;
  for (int t = wave; t < 56; t += 8) {
    int tr = t / 4, tc = t % 4;
    v8f c = {0.f, 0.f, 0.f, 0.f, 0.f, 0.f, 0.f, 0.f};
    for (int ks0 = 0; ks0 < NP; ks0 += 32) {
      U16 a, b;
      const float* prow = S + (size_t)(tr * 16 + lrow) * NP + ks0 + halfsel * 8;
#pragma unroll
      for (int j = 0; j < 8; ++j) a.s[j] = f2bf(prow[j]);
#pragma unroll
      for (int j = 0; j < 8; ++j) a.s[8 + j] = f2bf(prow[16 + j]);
      const unsigned short* vp = vts + (size_t)(tc * 16 + lrow) * NP + ks0 + halfsel * 8;
      b.q[0] = *(const u32x4*)(vp);
      b.q[1] = *(const u32x4*)(vp + 16);
      c = __builtin_amdgcn_wmma_f32_16x16x32_bf16(false, a.v, false, b.v,
                                                  (short)0, c, false, false);
    }
#pragma unroll
    for (int r = 0; r < 8; ++r) {
      int m = tr * 16 + r + halfsel * 8;
      if (m < 196)
        attn_out[(size_t)(win * NTOK + m) * C_ + head * HD + tc * 16 + lrow] = f2bf(c[r]);
    }
  }
}

// ---------------------------------------------------------------------------
// FRM helper kernels
// ---------------------------------------------------------------------------
__global__ __launch_bounds__(256) void avgmax_kernel(const float* __restrict__ rgb,
                                                     const float* __restrict__ xf,
                                                     float* __restrict__ yvec) {
  __shared__ float rsum[256], rmax[256];
  int blk = blockIdx.x;                   // 4 * 2 * 768
  int bb = blk / 1536, rem = blk % 1536, tensor = rem / C_, c = rem % C_;
  const float* src = tensor ? xf : rgb;
  float s = 0.f, m = -3.4e38f;
  for (int i = threadIdx.x; i < HW * HW; i += 256) {
    float v = src[((size_t)bb * HW * HW + i) * C_ + c];
    s += v; m = fmaxf(m, v);
  }
  rsum[threadIdx.x] = s; rmax[threadIdx.x] = m; __syncthreads();
  for (int off = 128; off > 0; off >>= 1) {
    if (threadIdx.x < off) {
      rsum[threadIdx.x] += rsum[threadIdx.x + off];
      rmax[threadIdx.x] = fmaxf(rmax[threadIdx.x], rmax[threadIdx.x + off]);
    }
    __syncthreads();
  }
  if (threadIdx.x == 0) {
    yvec[bb * 3072 + tensor * C_ + c] = rsum[0] * (1.0f / 4096.0f);
    yvec[bb * 3072 + 1536 + tensor * C_ + c] = rmax[0];
  }
}

__global__ __launch_bounds__(256) void cw1_kernel(const float* __restrict__ y,
                                                  const float* __restrict__ w,
                                                  const float* __restrict__ b,
                                                  float* __restrict__ t1) {
  int i = blockIdx.x * 256 + threadIdx.x;
  if (i >= B_ * 3072) return;
  int bb = i / 3072, o = i % 3072;
  const float* yr = y + (size_t)bb * 3072;
  const float* wr = w + (size_t)o * 3072;
  float acc = b[o];
  for (int k = 0; k < 3072; ++k) acc += yr[k] * wr[k];
  t1[i] = acc > 0.f ? acc : 0.f;
}

__global__ __launch_bounds__(256) void cw2_kernel(const float* __restrict__ t1,
                                                  const float* __restrict__ w,
                                                  const float* __restrict__ b,
                                                  float* __restrict__ cwv) {
  int i = blockIdx.x * 256 + threadIdx.x;
  if (i >= B_ * 1536) return;
  int bb = i / 1536, o = i % 1536;
  const float* tr = t1 + (size_t)bb * 3072;
  const float* wr = w + (size_t)o * 3072;
  float acc = b[o];
  for (int k = 0; k < 3072; ++k) acc += tr[k] * wr[k];
  cwv[i] = 1.0f / (1.0f + __expf(-acc));
}

__global__ __launch_bounds__(256) void pack_kernel(const float* __restrict__ rgb,
                                                   const float* __restrict__ xf,
                                                   unsigned short* __restrict__ afeat) {
  size_t i = (size_t)blockIdx.x * 256 + threadIdx.x;
  if (i >= (size_t)MT * C_) return;
  size_t token = i / C_;
  int c = (int)(i % C_);
  afeat[token * (2 * C_) + c]      = f2bf(rgb[i]);
  afeat[token * (2 * C_) + C_ + c] = f2bf(xf[i]);
}

__global__ __launch_bounds__(256) void sw2_kernel(const unsigned short* __restrict__ srelu,
                                                  const float* __restrict__ w,
                                                  const float* __restrict__ b,
                                                  float* __restrict__ s2) {
  int t = blockIdx.x * 256 + threadIdx.x;
  if (t >= MT) return;
  const unsigned short* sr = srelu + (size_t)t * C_;
  float a0 = b[0], a1 = b[1];
  for (int k = 0; k < C_; ++k) {
    float v = bf2f(sr[k]);
    a0 += v * w[k];
    a1 += v * w[C_ + k];
  }
  s2[t * 2 + 0] = 1.0f / (1.0f + __expf(-a0));
  s2[t * 2 + 1] = 1.0f / (1.0f + __expf(-a1));
}

__global__ __launch_bounds__(256) void final_kernel(const float* __restrict__ rgb,
                                                    const float* __restrict__ xf,
                                                    const float* __restrict__ cwv,
                                                    const float* __restrict__ s2,
                                                    float* __restrict__ out) {
  size_t i = (size_t)blockIdx.x * 256 + threadIdx.x;
  if (i >= (size_t)MT * C_) return;
  size_t token = i / C_;
  int c = (int)(i % C_);
  int bb = (int)(token >> 12);
  float x1 = rgb[i], x2 = xf[i];
  float c0 = cwv[bb * 1536 + c], c1 = cwv[bb * 1536 + C_ + c];
  float s0 = s2[token * 2 + 0], s1 = s2[token * 2 + 1];
  out[i] = x1 + 0.5f * (c1 + s1) * x2;                 // rgb_out
  out[(size_t)MT * C_ + i] = x2 + 0.5f * (c0 + s0) * x1; // x_out
}

// ---------------------------------------------------------------------------
// Launcher
// ---------------------------------------------------------------------------
extern "C" void kernel_launch(void* const* d_in, const int* in_sizes, int n_in,
                              void* d_out, int out_size, void* d_ws, size_t ws_size,
                              hipStream_t stream) {
  (void)in_sizes; (void)n_in; (void)out_size; (void)ws_size;
  const float* rgb     = (const float*)d_in[0];
  const float* xemb    = (const float*)d_in[1];
  const float* norm1_w = (const float*)d_in[2];
  const float* norm1_b = (const float*)d_in[3];
  const float* qkv_w   = (const float*)d_in[4];
  const float* qkv_b   = (const float*)d_in[5];
  const float* rel_h   = (const float*)d_in[6];
  const float* rel_w   = (const float*)d_in[7];
  const float* proj_w  = (const float*)d_in[8];
  const float* proj_b  = (const float*)d_in[9];
  const float* norm2_w = (const float*)d_in[10];
  const float* norm2_b = (const float*)d_in[11];
  const float* fc1_w   = (const float*)d_in[12];
  const float* fc1_b   = (const float*)d_in[13];
  const float* fc2_w   = (const float*)d_in[14];
  const float* fc2_b   = (const float*)d_in[15];
  const float* cw1_w   = (const float*)d_in[16];
  const float* cw1_b   = (const float*)d_in[17];
  const float* cw2_w   = (const float*)d_in[18];
  const float* cw2_b   = (const float*)d_in[19];
  const float* sw1_w   = (const float*)d_in[20];
  const float* sw1_b   = (const float*)d_in[21];
  const float* sw2_w   = (const float*)d_in[22];
  const float* sw2_b   = (const float*)d_in[23];

  char* ws = (char*)d_ws;
  unsigned short* wqkv  = (unsigned short*)(ws + OFF_WQKV);
  unsigned short* wproj = (unsigned short*)(ws + OFF_WPROJ);
  unsigned short* wfc1  = (unsigned short*)(ws + OFF_WFC1);
  unsigned short* wfc2  = (unsigned short*)(ws + OFF_WFC2);
  unsigned short* wsw1  = (unsigned short*)(ws + OFF_WSW1);
  unsigned short* qb    = (unsigned short*)(ws + OFF_Q);
  unsigned short* kb    = (unsigned short*)(ws + OFF_K);
  unsigned short* vb    = (unsigned short*)(ws + OFF_V);
  unsigned short* afeat = (unsigned short*)(ws + OFF_AFEAT);
  unsigned short* awin  = (unsigned short*)(ws + OFF_AWIN);
  unsigned short* aout  = (unsigned short*)(ws + OFF_AOUT);
  unsigned short* hln2  = (unsigned short*)(ws + OFF_HLN2);
  unsigned short* hmid  = (unsigned short*)(ws + OFF_HMID);
  unsigned short* srelu = (unsigned short*)(ws + OFF_SRELU);
  float* xattn = (float*)(ws + OFF_XATTN);
  float* xfin  = (float*)(ws + OFF_XFIN);
  float* yvec  = (float*)(ws + OFF_YVEC);
  float* t1    = (float*)(ws + OFF_T1);
  float* cwv   = (float*)(ws + OFF_CWV);
  float* s2    = (float*)(ws + OFF_S2);

  // zero padded rows of q/k/v and padded window slots of A_win (capture-safe)
  hipMemsetAsync(ws + OFF_Q, 0, SZ_QKV1 * 3, stream);
  hipMemsetAsync(ws + OFF_AWIN, 0, SZ_AWIN, stream);

  auto cvt = [&](const float* in, unsigned short* out, int n) {
    cvt_bf16<<<(n + 255) / 256, 256, 0, stream>>>(in, out, n);
  };
  cvt(qkv_w, wqkv, 3 * C_ * C_);
  cvt(proj_w, wproj, C_ * C_);
  cvt(fc1_w, wfc1, DFF * C_);
  cvt(fc2_w, wfc2, C_ * DFF);
  cvt(sw1_w, wsw1, C_ * 2 * C_);

  // LN1 -> window-gathered bf16 A
  ln_kernel<<<MT, 256, 0, stream>>>(xemb, norm1_w, norm1_b, awin, 1);

  // QKV GEMM (19600 x 2304 x 768) with q/k/v scatter epilogue
  {
    EpiParams p{qkv_b, nullptr, nullptr, nullptr, qb, kb, vb};
    gemm_bf16<0><<<dim3(2304 / 128, (MW + 127) / 128), 256, 0, stream>>>(
        awin, wqkv, MW, 2304, C_, p);
  }

  // fused windowed attention, 1200 (window,head) blocks, ~294 KB dynamic LDS
  hipFuncSetAttribute((const void*)attn_kernel,
                      hipFuncAttributeMaxDynamicSharedMemorySize, ATTN_SMEM);
  attn_kernel<<<NWIN * NHED, 256, ATTN_SMEM, stream>>>(qb, kb, vb, rel_h, rel_w, aout);

  // proj GEMM + shortcut, scatter back to (B,H,W,C)
  {
    EpiParams p{proj_b, xemb, xattn, nullptr, nullptr, nullptr, nullptr};
    gemm_bf16<1><<<dim3(C_ / 128, (MW + 127) / 128), 256, 0, stream>>>(
        aout, wproj, MW, C_, C_, p);
  }

  // LN2, fc1 (+GELU), fc2 (+residual)
  ln_kernel<<<MT, 256, 0, stream>>>(xattn, norm2_w, norm2_b, hln2, 0);
  {
    EpiParams p{fc1_b, nullptr, nullptr, hmid, nullptr, nullptr, nullptr};
    gemm_bf16<2><<<dim3(DFF / 128, MT / 128), 256, 0, stream>>>(
        hln2, wfc1, MT, DFF, C_, p);
  }
  {
    EpiParams p{fc2_b, xattn, xfin, nullptr, nullptr, nullptr, nullptr};
    gemm_bf16<3><<<dim3(C_ / 128, MT / 128), 256, 0, stream>>>(
        hmid, wfc2, MT, C_, DFF, p);
  }

  // FRM
  avgmax_kernel<<<B_ * 2 * C_, 256, 0, stream>>>(rgb, xfin, yvec);
  cw1_kernel<<<(B_ * 3072 + 255) / 256, 256, 0, stream>>>(yvec, cw1_w, cw1_b, t1);
  cw2_kernel<<<(B_ * 1536 + 255) / 256, 256, 0, stream>>>(t1, cw2_w, cw2_b, cwv);
  pack_kernel<<<(int)(((size_t)MT * C_ + 255) / 256), 256, 0, stream>>>(rgb, xfin, afeat);
  {
    EpiParams p{sw1_b, nullptr, nullptr, srelu, nullptr, nullptr, nullptr};
    gemm_bf16<4><<<dim3(C_ / 128, MT / 128), 256, 0, stream>>>(
        afeat, wsw1, MT, C_, 2 * C_, p);
  }
  sw2_kernel<<<(MT + 255) / 256, 256, 0, stream>>>(srelu, sw2_w, sw2_b, s2);
  final_kernel<<<(int)(((size_t)MT * C_ + 255) / 256), 256, 0, stream>>>(
      rgb, xfin, cwv, s2, (float*)d_out);
}
